// Seq2SeqDecoder_26439818674412
// MI455X (gfx1250) — matrix operand order repaired and checked by reading, not compiled, not run on hardware
//
#include <hip/hip_runtime.h>
#include <hip/hip_bf16.h>

// Problem constants (from reference)
#define Vv 32000
#define Dd 512
#define Hh 512
#define Ll 4
#define Bb 16
#define Tt 128
#define NSUB 16              // sub-workgroups (WGPs) per LSTM layer
#define SUBCOLS 32           // hidden columns owned per sub-workgroup

typedef __attribute__((ext_vector_type(16))) _Float16 v16h;
typedef __attribute__((ext_vector_type(8)))  float    v8f;

__device__ __forceinline__ v8f wmma16(v16h a, v16h b, v8f c) {
    // D = A(16x32 f16) * B(32x16 f16) + C(16x16 f32)
    return __builtin_amdgcn_wmma_f32_16x16x32_f16(
        false, a, false, b, (short)0, c, false, false);
}

// ---- CDNA5 async global->LDS (ASYNCcnt) --------------------------------
// LDS byte offset = low 32 bits of the generic pointer.
__device__ __forceinline__ void async_copy_b128(void* lds_dst, const void* gsrc) {
    unsigned ldsa = (unsigned)(uintptr_t)lds_dst;
    asm volatile("global_load_async_to_lds_b128 %0, %1, off"
                 :: "v"(ldsa), "v"(gsrc) : "memory");
}
__device__ __forceinline__ void wait_async0() {
    asm volatile("s_wait_asynccnt 0x0" ::: "memory");
}

// A-fragment (16x32, f16) from a row-major [16 x 512] tile (LDS or global).
// Contiguous k order used here (same as B): lanes 0-15 -> row M=lane, halves = K {k0..k0+15}
//               lanes 16-31 -> row M=lane-16, halves = K {k0+16..k0+31}
__device__ __forceinline__ v16h load_a_16x512(const _Float16* base, int k0, int lane) {
    int row = lane & 15;
    int kk  = k0 + ((lane & 16) ? 16 : 0);
    union { v16h v; uint4 q[2]; } u;
    u.q[0] = *(const uint4*)(base + row * Hh + kk);
    u.q[1] = *(const uint4*)(base + row * Hh + kk + 8);
    return u.v;
}

// B-fragment (32x16, f16) where B[k][n] = W[n][k], W row-major [N x 512] in global/L2.
// Contiguous k order used here (same as A): lanes 0-15 -> col N=n0+lane, halves = K {k0..k0+15}
//               lanes 16-31 -> col N=n0+lane-16, halves = K {k0+16..k0+31}
__device__ __forceinline__ v16h load_b_w(const _Float16* W, size_t n0, int k0, int lane) {
    size_t col = n0 + (size_t)(lane & 15);
    int kk = k0 + ((lane & 16) ? 16 : 0);
    union { v16h v; uint4 q[2]; } u;
    u.q[0] = *(const uint4*)(W + col * Hh + kk);
    u.q[1] = *(const uint4*)(W + col * Hh + kk + 8);
    return u.v;
}

// B-fragment from LDS tile stored in fragment order:
// tile = 16 k-slabs of 1KB; slab s: lane's q0 at s*1024 + lane*16, q1 at +512.
__device__ __forceinline__ v16h load_b_frag(const _Float16* tile, int s, int lane) {
    const char* slab = (const char*)tile + s * 1024;
    union { v16h v; uint4 q[2]; } u;
    u.q[0] = *(const uint4*)(slab + lane * 16);
    u.q[1] = *(const uint4*)(slab + 512 + lane * 16);
    return u.v;
}

__device__ __forceinline__ float sigmoidf(float x) {
    return 1.0f / (1.0f + __expf(-x));
}

// ---------------------------------------------------------------- f32 -> f16
__global__ void cvt_f32_f16(const float* __restrict__ src, _Float16* __restrict__ dst, size_t n) {
    size_t i = (size_t)blockIdx.x * blockDim.x + threadIdx.x;
    size_t stride = (size_t)gridDim.x * blockDim.x;
    for (; i < n; i += stride) dst[i] = (_Float16)src[i];
}

// ------------------------------------------------- embedding + relu -> f16 x0
// x0 layout: [T][B][H] halves (row = t*16 + b). Also zero the done counters.
__global__ void embed_relu(const int* __restrict__ dec, const float* __restrict__ emb,
                           _Float16* __restrict__ x0, int* __restrict__ done) {
    if (blockIdx.x == 0)
        for (int i = threadIdx.x; i < Ll * Tt; i += blockDim.x) done[i] = 0;
    int row = blockIdx.x;            // 0..B*T-1, row = t*16 + b
    int t = row >> 4, b = row & 15;
    int id = dec[b * Tt + t];        // dec_input is [B, T]
    const float* src = emb + (size_t)id * Dd;
    _Float16* d = x0 + (size_t)row * Dd;
    for (int k = threadIdx.x; k < Dd; k += blockDim.x) {
        float v = src[k];
        d[k] = (_Float16)(v > 0.0f ? v : 0.0f);
    }
}

// --------------------------------------------------------- persistent LSTM
// grid = L*NSUB workgroups; block = 128 threads (4 waves); ~305KB dynamic LDS.
// Each workgroup owns SUBCOLS=32 hidden columns of one layer; its 256KB weight
// slice (2 mats x 4 gates x 32 cols x 512 K, f16) is swizzled into LDS once in
// WMMA B-fragment order and stays resident for all T steps.
// Handoff: done[l*T+t] counts subs of layer l that finished step t.
//   layer l step t needs: x_t   -> (l==0) or done[(l-1)*T+t]==NSUB
//                         h_t-1 -> (t==0) or done[l*T+t-1]==NSUB
// h published to global hbuf[l][t&1] (parity double buffer).
#define WLDS_BYTES   (16 * 16384)                 // 262144: 16 tiles x 16KB
#define XLDS_OFF     (WLDS_BYTES)                 // +16384  x_t   (16x512 f16)
#define HLDS_OFF     (XLDS_OFF + 16384)           // +16384  h_t-1 (16x512 f16)
#define GBUF_OFF     (HLDS_OFF + 16384)           // +8192   gate tiles (8 x 16x16 f32)
#define CST_OFF      (GBUF_OFF + 8192)            // +2048   c-state (16x32 f32)
#define BLDS_OFF     (CST_OFF + 2048)             // +512    combined bias (4x32 f32)
#define LSTM_SMEM    (BLDS_OFF + 512)             // 305,664 bytes

__global__ __launch_bounds__(128) void lstm_persistent(
    _Float16* xbuf,                   // (L+1) slabs of [T][16][512] f16
    const _Float16* __restrict__ Wih, const _Float16* __restrict__ Whh,
    const float* __restrict__ bih, const float* __restrict__ bhh,
    const float* __restrict__ h0g, const float* __restrict__ c0g,
    _Float16* hbuf,                   // [L][2][16][512] f16 (parity buffers)
    float* __restrict__ outHC,        // hT [L,B,H] then cT [L,B,H]
    int* done)
{
    extern __shared__ char smem[];
    _Float16* wlds = (_Float16*)smem;
    _Float16* xlds = (_Float16*)(smem + XLDS_OFF);
    _Float16* hlds = (_Float16*)(smem + HLDS_OFF);
    float*    gbuf = (float*)(smem + GBUF_OFF);
    float*    cst  = (float*)(smem + CST_OFF);
    float*    blds = (float*)(smem + BLDS_OFF);

    const int l    = blockIdx.x >> 4;
    const int sub  = blockIdx.x & 15;
    const int tid  = threadIdx.x;
    const int wave = tid >> 5;
    const int lane = tid & 31;

    const _Float16* Wihl = Wih + (size_t)l * 4 * Hh * Hh;
    const _Float16* Whhl = Whh + (size_t)l * 4 * Hh * Hh;
    const float* bihl = bih + (size_t)l * 4 * Hh;
    const float* bhhl = bhh + (size_t)l * 4 * Hh;

    // ---- one-time: swizzle weight slice into LDS in B-fragment order ----
    // tile idx = ((mat*4 + gate)*2 + chunk); cols = gate*512 + sub*32 + chunk*16
    for (int tile = 0; tile < 16; ++tile) {
        int mat  = tile >> 3;
        int gate = (tile >> 1) & 3;
        int c    = tile & 1;
        const _Float16* Wsrc = mat ? Whhl : Wihl;
        size_t colbase = (size_t)gate * Hh + sub * SUBCOLS + c * 16;
        _Float16* dst = wlds + (size_t)tile * 8192;
        for (int slot = tid; slot < 512; slot += 128) {
            int s = slot >> 5, ls = slot & 31;            // k-step, pseudo-lane
            size_t col = colbase + (ls & 15);
            int kk = s * 32 + ((ls & 16) ? 16 : 0);
            uint4 q0 = *(const uint4*)(Wsrc + col * Hh + kk);
            uint4 q1 = *(const uint4*)(Wsrc + col * Hh + kk + 8);
            char* slab = (char*)dst + s * 1024;
            *(uint4*)(slab + ls * 16) = q0;
            *(uint4*)(slab + 512 + ls * 16) = q1;
        }
    }
    // combined bias for this sub's 32 cols, and initial c-state
    if (tid < 128) {
        int g = tid >> 5, jj = tid & 31;
        blds[tid] = bihl[g * Hh + sub * SUBCOLS + jj] + bhhl[g * Hh + sub * SUBCOLS + jj];
    }
    for (int e = tid; e < Bb * SUBCOLS; e += 128) {
        int m = e >> 5, jj = e & 31;
        cst[m * SUBCOLS + jj] = c0g[(size_t)l * Bb * Hh + m * Hh + sub * SUBCOLS + jj];
    }
    __syncthreads();

    // wave roles: c = chunk (0/1), gate pair p -> gates {2p, 2p+1}
    const int c  = wave >> 1;
    const int p  = wave & 1;
    const int g0 = p * 2, g1 = p * 2 + 1;
    const _Float16* tx0 = wlds + (size_t)((0 * 4 + g0) * 2 + c) * 8192;
    const _Float16* th0 = wlds + (size_t)((1 * 4 + g0) * 2 + c) * 8192;
    const _Float16* tx1 = wlds + (size_t)((0 * 4 + g1) * 2 + c) * 8192;
    const _Float16* th1 = wlds + (size_t)((1 * 4 + g1) * 2 + c) * 8192;

    for (int t = 0; t < Tt; ++t) {
        // ---- x dependency (acquire), then start async x staging ----
        if (l > 0) {
            const int idx = (l - 1) * Tt + t;
            while (__hip_atomic_load(&done[idx], __ATOMIC_ACQUIRE,
                                     __HIP_MEMORY_SCOPE_AGENT) < NSUB)
                __builtin_amdgcn_s_sleep(1);
        }
        const _Float16* xsrc = xbuf + ((size_t)l * Tt + t) * Bb * Hh;
        for (int i = tid * 8; i < Bb * Hh; i += 128 * 8)
            async_copy_b128(xlds + i, xsrc + i);   // overlaps with h spin below

        // ---- h dependency (acquire), then async h staging ----
        if (t > 0) {
            const int idx = l * Tt + (t - 1);
            while (__hip_atomic_load(&done[idx], __ATOMIC_ACQUIRE,
                                     __HIP_MEMORY_SCOPE_AGENT) < NSUB)
                __builtin_amdgcn_s_sleep(1);
            const _Float16* hsrc = hbuf + ((size_t)l * 2 + ((t + 1) & 1)) * Bb * Hh;
            for (int i = tid * 8; i < Bb * Hh; i += 128 * 8)
                async_copy_b128(hlds + i, hsrc + i);
        } else {
            const float* hsrc = h0g + (size_t)l * Bb * Hh;
            for (int i = tid; i < Bb * Hh; i += 128) hlds[i] = (_Float16)hsrc[i];
        }
        wait_async0();
        __syncthreads();

        // ---- gate GEMM: LDS-resident weights, shared A across gate pair ----
        v8f a0 = {}, a1 = {};
        for (int s = 0; s < 16; ++s) {
            int k0 = s * 32;
            v16h ax = load_a_16x512(xlds, k0, lane);
            v16h ah = load_a_16x512(hlds, k0, lane);
            a0 = wmma16(ax, load_b_frag(tx0, s, lane), a0);
            a0 = wmma16(ah, load_b_frag(th0, s, lane), a0);
            a1 = wmma16(ax, load_b_frag(tx1, s, lane), a1);
            a1 = wmma16(ah, load_b_frag(th1, s, lane), a1);
        }
        {
            int jl = lane & 15, mb = (lane & 16) ? 8 : 0;
#pragma unroll
            for (int r = 0; r < 8; ++r) {
                gbuf[(c * 4 + g0) * 256 + (mb + r) * 16 + jl] = a0[r];
                gbuf[(c * 4 + g1) * 256 + (mb + r) * 16 + jl] = a1[r];
            }
        }
        __syncthreads();

        // ---- fused cell math (fp32), c-state resident in LDS ----
        _Float16* hw = hbuf + ((size_t)l * 2 + (t & 1)) * Bb * Hh;
        for (int e = tid; e < Bb * SUBCOLS; e += 128) {
            int m = e >> 5, jj = e & 31;
            int cc = jj >> 4, jl = jj & 15;
            float gi = gbuf[(cc * 4 + 0) * 256 + m * 16 + jl] + blds[0 * 32 + jj];
            float gf = gbuf[(cc * 4 + 1) * 256 + m * 16 + jl] + blds[1 * 32 + jj];
            float gg = gbuf[(cc * 4 + 2) * 256 + m * 16 + jl] + blds[2 * 32 + jj];
            float go = gbuf[(cc * 4 + 3) * 256 + m * 16 + jl] + blds[3 * 32 + jj];
            float c_new = sigmoidf(gf) * cst[m * SUBCOLS + jj]
                        + sigmoidf(gi) * tanhf(gg);
            cst[m * SUBCOLS + jj] = c_new;
            float h_new = sigmoidf(go) * tanhf(c_new);
            int col = sub * SUBCOLS + jj;
            hw[(size_t)m * Hh + col] = (_Float16)h_new;
            float xo = h_new + (float)xlds[m * Hh + col];    // residual
            xbuf[(((size_t)(l + 1) * Tt + t) * Bb + m) * Hh + col] = (_Float16)xo;
            if (t == Tt - 1) {
                outHC[(size_t)l * Bb * Hh + m * Hh + col] = h_new;                      // hT
                outHC[(size_t)Ll * Bb * Hh + (size_t)l * Bb * Hh + m * Hh + col] = c_new; // cT
            }
        }

        __threadfence();   // publish h slice + x_out before counter bump
        __syncthreads();
        if (tid == 0)
            __hip_atomic_fetch_add(&done[l * Tt + t], 1, __ATOMIC_RELEASE,
                                   __HIP_MEMORY_SCOPE_AGENT);
    }
}

// --------------------------------------------------------- classifier GEMM
// out[b,t,v] = ys[t*16+b,:] @ W[v,:]^T + bias[v]
// Block: 64 rows (A staged in 64KB LDS via async-to-LDS) x 256 cols.
// 8 waves: each wave owns 2 N-tiles x 4 M-tiles (8 fp32 accumulator tiles).
__global__ __launch_bounds__(256) void cls_gemm(
    const _Float16* __restrict__ ys,   // [T*B, 512] rows = t*16+b
    const _Float16* __restrict__ W16,  // [V, 512]
    const float* __restrict__ bias,    // [V]
    float* __restrict__ out)           // [B, T, V]
{
    __shared__ __align__(16) _Float16 ablk[64 * Hh];   // 64 KB

    const int tid  = threadIdx.x;
    const int wave = tid >> 5;
    const int lane = tid & 31;
    const int m0 = blockIdx.y * 64;
    const size_t n0g = (size_t)blockIdx.x * 256;

    for (int i = tid * 8; i < 64 * Hh; i += 256 * 8)
        async_copy_b128(ablk + i, ys + (size_t)m0 * Hh + i);
    wait_async0();
    __syncthreads();

    const size_t n0 = n0g + (size_t)wave * 32;   // 2 N-tiles per wave
    v8f acc[4][2] = {};

    for (int k0 = 0; k0 < Hh; k0 += 32) {
        v16h b0 = load_b_w(W16, n0,      k0, lane);
        v16h b1 = load_b_w(W16, n0 + 16, k0, lane);
#pragma unroll
        for (int mt = 0; mt < 4; ++mt) {
            v16h a = load_a_16x512(ablk + mt * 16 * Hh, k0, lane);
            acc[mt][0] = wmma16(a, b0, acc[mt][0]);
            acc[mt][1] = wmma16(a, b1, acc[mt][1]);
        }
    }

    const int mbase = (lane & 16) ? 8 : 0;
#pragma unroll
    for (int nt = 0; nt < 2; ++nt) {
        size_t n = n0 + nt * 16 + (lane & 15);
        float bv = bias[n];
#pragma unroll
        for (int mt = 0; mt < 4; ++mt) {
#pragma unroll
            for (int r = 0; r < 8; ++r) {
                int m = m0 + mt * 16 + mbase + r;   // row in (t*16+b) order
                int t = m >> 4, b = m & 15;
                out[((size_t)b * Tt + t) * Vv + n] = acc[mt][nt][r] + bv;
            }
        }
    }
}

// ----------------------------------------------------------------- launcher
extern "C" void kernel_launch(void* const* d_in, const int* in_sizes, int n_in,
                              void* d_out, int out_size, void* d_ws, size_t ws_size,
                              hipStream_t stream) {
    const int*   dec  = (const int*)  d_in[0];
    const float* h0   = (const float*)d_in[1];
    const float* c0   = (const float*)d_in[2];
    const float* emb  = (const float*)d_in[3];
    const float* Wih  = (const float*)d_in[4];
    const float* Whh  = (const float*)d_in[5];
    const float* bih  = (const float*)d_in[6];
    const float* bhh  = (const float*)d_in[7];
    const float* Wcls = (const float*)d_in[8];
    const float* bcls = (const float*)d_in[9];
    float* out = (float*)d_out;

    // workspace layout (halves)
    const size_t XBUF_N = (size_t)(Ll + 1) * Tt * Bb * Hh;   // 5,242,880
    const size_t WIH_N  = (size_t)Ll * 4 * Hh * Hh;          // 4,194,304
    const size_t WCLS_N = (size_t)Vv * Hh;                   // 16,384,000
    const size_t HBUF_N = (size_t)Ll * 2 * Bb * Hh;          // 65,536
    _Float16* xbuf   = (_Float16*)d_ws;
    _Float16* Wih16  = xbuf   + XBUF_N;
    _Float16* Whh16  = Wih16  + WIH_N;
    _Float16* Wcls16 = Whh16  + WIH_N;
    _Float16* hbuf   = Wcls16 + WCLS_N;
    int*      done   = (int*)(hbuf + HBUF_N);                // L*T ints

    // 1) convert weights to f16 (idempotent, deterministic)
    cvt_f32_f16<<<2048, 256, 0, stream>>>(Wih,  Wih16,  WIH_N);
    cvt_f32_f16<<<2048, 256, 0, stream>>>(Whh,  Whh16,  WIH_N);
    cvt_f32_f16<<<4096, 256, 0, stream>>>(Wcls, Wcls16, WCLS_N);

    // 2) embedding + relu -> f16 [T,B,H]; also zero done counters
    embed_relu<<<Bb * Tt, 64, 0, stream>>>(dec, emb, xbuf, done);

    // 3) persistent pipelined LSTM: 16 WGPs per layer, LDS-resident weights
    const size_t LOGITS_N = (size_t)Bb * Tt * Vv;
    lstm_persistent<<<Ll * NSUB, 128, LSTM_SMEM, stream>>>(
        xbuf, Wih16, Whh16, bih, bhh, h0, c0, hbuf, out + LOGITS_N, done);

    // 4) classifier GEMM on ys = xbuf slab L
    dim3 cgrid(Vv / 256, (Bb * Tt) / 64);   // 125 x 32
    cls_gemm<<<cgrid, 256, 0, stream>>>(xbuf + (size_t)Ll * Tt * Bb * Hh,
                                        Wcls16, bcls, out);
}
